// ourRNNModel_6906307412188
// MI455X (gfx1250) — matrix operand order repaired and checked by reading, not compiled, not run on hardware
//
#include <hip/hip_runtime.h>
#include <hip/hip_bf16.h>

typedef __attribute__((ext_vector_type(16))) _Float16 v16h;
typedef __attribute__((ext_vector_type(8)))  float    v8f;

#define BB 256
#define TT 1024
#define DD 64
#define HH 128

// CDNA5 has V_TANH_F32; use the hardware trans op when the builtin exists.
#if __has_builtin(__builtin_amdgcn_tanhf)
__device__ __forceinline__ float fast_tanh(float x) { return __builtin_amdgcn_tanhf(x); }
#else
__device__ __forceinline__ float fast_tanh(float x) { return tanhf(x); }
#endif
// sigmoid(x) = 0.5*tanh(x/2) + 0.5  (1 trans op instead of exp+rcp)
__device__ __forceinline__ float fast_sigmoid(float x) {
  return __builtin_fmaf(0.5f, fast_tanh(0.5f * x), 0.5f);
}

// B fragment (32x16 f16): 16 consecutive K values from one weight row.
__device__ __forceinline__ v16h load_b_frag(const float* __restrict__ p) {
  float4 q0 = *(const float4*)(p + 0);
  float4 q1 = *(const float4*)(p + 4);
  float4 q2 = *(const float4*)(p + 8);
  float4 q3 = *(const float4*)(p + 12);
  v16h b;
  b[0]  = (_Float16)q0.x; b[1]  = (_Float16)q0.y; b[2]  = (_Float16)q0.z; b[3]  = (_Float16)q0.w;
  b[4]  = (_Float16)q1.x; b[5]  = (_Float16)q1.y; b[6]  = (_Float16)q1.z; b[7]  = (_Float16)q1.w;
  b[8]  = (_Float16)q2.x; b[9]  = (_Float16)q2.y; b[10] = (_Float16)q2.z; b[11] = (_Float16)q2.w;
  b[12] = (_Float16)q3.x; b[13] = (_Float16)q3.y; b[14] = (_Float16)q3.z; b[15] = (_Float16)q3.w;
  return b;
}

// A fragment (16x32 f16): elements 0..15 <- p[0..15]
__device__ __forceinline__ v16h load_a_frag(const float* __restrict__ p) {
  float4 q0 = *(const float4*)(p + 0);
  float4 q1 = *(const float4*)(p + 4);
  float4 q2 = *(const float4*)(p + 8);
  float4 q3 = *(const float4*)(p + 12);
  v16h a;
  a[0]  = (_Float16)q0.x; a[1]  = (_Float16)q0.y; a[2]  = (_Float16)q0.z; a[3]  = (_Float16)q0.w;
  a[4]  = (_Float16)q1.x; a[5]  = (_Float16)q1.y; a[6]  = (_Float16)q1.z; a[7]  = (_Float16)q1.w;
  a[8]  = (_Float16)q2.x; a[9]  = (_Float16)q2.y; a[10] = (_Float16)q2.z; a[11] = (_Float16)q2.w;
  a[12] = (_Float16)q3.x; a[13] = (_Float16)q3.y; a[14] = (_Float16)q3.z; a[15] = (_Float16)q3.w;
  return a;
}

// -------- Phase 1: gxK = x@WxK^T + b ; z = tanh(x@Wxz^T + b) --------------
// One block = 16 rows of (b,t); 8 waves, wave w owns h-tile [16w,16w+16) of
// BOTH weight matrices (shared A fragment, two accumulators).
__global__ __launch_bounds__(256) void rnn_proj(
    const float* __restrict__ x,
    const float* __restrict__ WxK, const float* __restrict__ bxK,
    const float* __restrict__ Wxz, const float* __restrict__ bxz,
    float* __restrict__ gxK_out, float* __restrict__ z_out) {
  const int lane = threadIdx.x & 31;
  const int wave = threadIdx.x >> 5;
  const int hi   = (lane >> 4) & 1;   // high half-wave
  const int m    = lane & 15;
  const int row_base = blockIdx.x << 4;
  const int h = (wave << 4) + m;

  v8f cK = {}; v8f cz = {};
#pragma unroll
  for (int ks = 0; ks < 2; ++ks) {           // K = 64 = 2 x 32
    const int k0 = ks << 5;
    v16h a  = load_a_frag(x   + (row_base + m) * DD + k0 + hi * 16);
    v16h bK = load_b_frag(WxK + h * DD + k0 + hi * 16);
    v16h bz = load_b_frag(Wxz + h * DD + k0 + hi * 16);
    cK = __builtin_amdgcn_wmma_f32_16x16x32_f16(false, a, false, bK, (short)0, cK, false, false);
    cz = __builtin_amdgcn_wmma_f32_16x16x32_f16(false, a, false, bz, (short)0, cz, false, false);
  }
  const float biasK = bxK[h];
  const float biasz = bxz[h];
#pragma unroll
  for (int r = 0; r < 8; ++r) {
    const int row = row_base + r + hi * 8;
    const int idx = row * HH + h;
    gxK_out[idx] = cK[r] + biasK;            // staged gxK (overwritten by scan)
    z_out[idx]   = fast_tanh(cz[r] + biasz); // staged z   (overwritten by scan)
  }
}

// -------- Phase 2: sequential scan over T ---------------------------------
// One block = 16 batch rows, 4 waves; wave w owns h-tiles {2w,2w+1}.
// W_hK held as f16 B-fragments in registers for the whole scan; h state
// double-buffered in LDS; gxK/z prefetched one timestep ahead.
__global__ __launch_bounds__(128) void rnn_scan(
    const float* __restrict__ WhK, const float* __restrict__ bhK,
    float* __restrict__ out0, float* __restrict__ out1) {
  __shared__ float hbuf[2][16 * HH];
  const int lane  = threadIdx.x & 31;
  const int wave  = threadIdx.x >> 5;
  const int hi    = (lane >> 4) & 1;
  const int m     = lane & 15;
  const int bbase = blockIdx.x << 4;

  for (int i = threadIdx.x; i < 16 * HH; i += 128) hbuf[0][i] = 0.0f;  // h0 = 0

  v16h  Bf[2][4];
  int   hcol[2];
  float bh[2];
#pragma unroll
  for (int tl = 0; tl < 2; ++tl) {
    const int h = wave * 32 + tl * 16 + m;
    hcol[tl] = h;
    bh[tl]   = bhK[h];
#pragma unroll
    for (int ks = 0; ks < 4; ++ks)           // K = 128 = 4 x 32
      Bf[tl][ks] = load_b_frag(WhK + h * HH + ks * 32 + hi * 16);
  }

  int rowc[2][8];
#pragma unroll
  for (int tl = 0; tl < 2; ++tl)
#pragma unroll
    for (int r = 0; r < 8; ++r)
      rowc[tl][r] = (bbase + r + hi * 8) * (TT * HH) + hcol[tl];

  // prefetch t = 0
  float gK[2][8], zv[2][8];
#pragma unroll
  for (int tl = 0; tl < 2; ++tl)
#pragma unroll
    for (int r = 0; r < 8; ++r) {
      gK[tl][r] = out0[rowc[tl][r]];
      zv[tl][r] = out1[rowc[tl][r]];
    }

  __syncthreads();

  int cur = 0;
  for (int t = 0; t < TT; ++t) {
    v8f c0 = {}; v8f c1 = {};
#pragma unroll
    for (int ks = 0; ks < 4; ++ks) {
      v16h a = load_a_frag(&hbuf[cur][m * HH + ks * 32 + hi * 16]);
      c0 = __builtin_amdgcn_wmma_f32_16x16x32_f16(false, a, false, Bf[0][ks], (short)0, c0, false, false);
      c1 = __builtin_amdgcn_wmma_f32_16x16x32_f16(false, a, false, Bf[1][ks], (short)0, c1, false, false);
    }
    const int nxt = cur ^ 1;
    const int tn  = (t < TT - 1) ? (t + 1) : (TT - 1);
#pragma unroll
    for (int tl = 0; tl < 2; ++tl) {
#pragma unroll
      for (int r = 0; r < 8; ++r) {
        const int   M    = r + hi * 8;
        const float hold = hbuf[cur][M * HH + hcol[tl]];
        const float gH   = (tl ? c1[r] : c0[r]) + bh[tl];
        const float kg   = fast_sigmoid(gK[tl][r] + gH);
        const float hn   = fast_tanh(__builtin_fmaf(kg, zv[tl][r] - hold, hold));
        hbuf[nxt][M * HH + hcol[tl]] = hn;
        const int idx = rowc[tl][r] + t * HH;
        out0[idx] = hn;                       // outs (copy 0)
        out1[idx] = hn;                       // outs (copy 1)
        // prefetch next timestep's staged gxK / z (consumed next iteration,
        // overlaps the next WMMA phase)
        const int nidx = rowc[tl][r] + tn * HH;
        gK[tl][r] = out0[nidx];
        zv[tl][r] = out1[nidx];
      }
    }
    cur = nxt;
    __syncthreads();
  }
}

extern "C" void kernel_launch(void* const* d_in, const int* in_sizes, int n_in,
                              void* d_out, int out_size, void* d_ws, size_t ws_size,
                              hipStream_t stream) {
  const float* x   = (const float*)d_in[0];
  const float* WxK = (const float*)d_in[1];
  const float* bxK = (const float*)d_in[2];
  const float* Wxz = (const float*)d_in[3];
  const float* bxz = (const float*)d_in[4];
  const float* WhK = (const float*)d_in[5];
  const float* bhK = (const float*)d_in[6];
  float* out0 = (float*)d_out;
  float* out1 = out0 + (size_t)BB * TT * HH;

  rnn_proj<<<(BB * TT) / 16, 256, 0, stream>>>(x, WxK, bxK, Wxz, bxz, out0, out1);
  rnn_scan<<<BB / 16, 128, 0, stream>>>(WhK, bhK, out0, out1);
}
